// CausalSelfAttention_8796093022599
// MI455X (gfx1250) — matrix operand order repaired and checked by reading, not compiled, not run on hardware
//
#include <hip/hip_runtime.h>
#include <hip/hip_bf16.h>
#include <stdint.h>

// ---------------------------------------------------------------------------
// CDNA5 (gfx1250) causal self-attention.
// All matmuls via v_wmma_f32_16x16x32_bf16 (f32 accumulate).
// Double-buffered LDS staging: next tile issued (global_load_async_to_lds_b64
// on ASYNCcnt + converting ds stores) before computing the current tile; one
// barrier per k-step.  Next-next tile prefetched via global_prefetch_b8 (WGP).
// Pipeline: gemm_qkv (f32 -> bf16 qkv) -> flash attention -> gemm_proj (-> f32).
// ---------------------------------------------------------------------------

typedef __attribute__((ext_vector_type(16))) __bf16 v16bf;
typedef __attribute__((ext_vector_type(8)))  float  v8f;

union FragB16 { v16bf v; uint32_t u[8]; };

#define NEG_BIG (-3.0e38f)
#define LDSTR   36   // GEMM LDS halfword stride (72B rows: 8B-aligned)
#define KSTR    68   // attention K-tile halfword stride (136B rows: 8B-aligned)

__device__ __forceinline__ uint32_t pk_bf16(float lo, float hi) {
  uint32_t a = __builtin_bit_cast(uint32_t, lo);
  uint32_t b = __builtin_bit_cast(uint32_t, hi);
  a += 0x7FFFu + ((a >> 16) & 1u);   // round-to-nearest-even
  b += 0x7FFFu + ((b >> 16) & 1u);
  return (a >> 16) | (b & 0xFFFF0000u);
}
__device__ __forceinline__ uint16_t bf16_1(float f) {
  uint32_t a = __builtin_bit_cast(uint32_t, f);
  a += 0x7FFFu + ((a >> 16) & 1u);
  return (uint16_t)(a >> 16);
}

// A-fragment k index for slot v (16-bit A 16x32 layout, ISA 7.12.2)
__device__ __forceinline__ int amap(int v, int lh) {
  return ((v & 4) ? 16 : 0) + lh * 8 + (v & 3) * 2;
}
// B-fragment k index for slot v (same k order as amap)
__device__ __forceinline__ int bmap(int v, int lh) {
  return ((v & 4) ? 16 : 0) + lh * 8 + (v & 3) * 2;
}

// Async copy 8 bytes Memory -> LDS.  LDS aperture flat addresses carry the LDS
// byte offset in addr[31:0], so truncating the generic pointer gives VDST.
__device__ __forceinline__ void async_b64(void* lds, const void* gsrc) {
  uint32_t l = (uint32_t)(uintptr_t)lds;
  asm volatile("global_load_async_to_lds_b64 %0, %1, off"
               :: "v"(l), "v"(gsrc) : "memory");
}
__device__ __forceinline__ void wait_async0() {
  asm volatile("s_wait_asynccnt 0" ::: "memory");
}

// ---------------------------------------------------------------------------
// GEMM 1: Out_bf16[M][N] = bf16( X_f32[M][K] @ W_f32[K][N] + bias )
// 256 threads, block tile 128(M) x 64(N), BK=32, double-buffered LDS.
// ---------------------------------------------------------------------------
__global__ __launch_bounds__(256) void gemm_qkv_kernel(
    const float* __restrict__ X, const float* __restrict__ W,
    const float* __restrict__ bias, uint16_t* __restrict__ Out,
    int M, int N, int K) {
  __shared__ uint16_t At[2][128 * LDSTR];   // [m][k] row-major
  __shared__ uint16_t Bt[2][64 * LDSTR];    // [n][k] transposed

  const int tid  = threadIdx.x;
  const int lane = tid & 31;
  const int wave = tid >> 5;
  const int lm = lane & 15, lh = lane >> 4;
  const int nbase = blockIdx.x * 64;
  const int mbase = blockIdx.y * 128;

  auto fill = [&](int buf, int kk) {
    { // A tile (f32 -> bf16): thread -> row tid>>1, 16 cols
      int row = tid >> 1, cs = (tid & 1) * 16;
      const float* src = X + (size_t)(mbase + row) * K + kk + cs;
      uint32_t* dst = (uint32_t*)(At[buf] + row * LDSTR + cs);
#pragma unroll
      for (int i = 0; i < 8; i++) dst[i] = pk_bf16(src[2 * i], src[2 * i + 1]);
      __builtin_prefetch(src + 32, 0, 3);               // tile after next
    }
    { // B tile transposed: thread -> krow tid>>3, 8 n's
      int kr = tid >> 3, ns = (tid & 7) * 8;
      const float* src = W + (size_t)(kk + kr) * N + nbase + ns;
#pragma unroll
      for (int i = 0; i < 8; i++) Bt[buf][(ns + i) * LDSTR + kr] = bf16_1(src[i]);
      __builtin_prefetch(src + (size_t)32 * N, 0, 3);
    }
  };

  v8f acc[4] = {};

  fill(0, 0);
  __syncthreads();

  for (int k0 = 0; k0 < K; k0 += 32) {
    const int cur = (k0 >> 5) & 1;
    if (k0 + 32 < K) fill(cur ^ 1, k0 + 32);   // overlap next fill with compute

    FragB16 a;
    {
      const uint16_t* base = At[cur] + (wave * 16 + lm) * LDSTR;
#pragma unroll
      for (int v = 0; v < 8; v++) a.u[v] = *(const uint32_t*)(base + amap(v, lh));
    }
#pragma unroll
    for (int nt = 0; nt < 4; nt++) {
      FragB16 b;
      const uint16_t* base = Bt[cur] + (nt * 16 + lm) * LDSTR;
#pragma unroll
      for (int v = 0; v < 8; v++) b.u[v] = *(const uint32_t*)(base + bmap(v, lh));
      acc[nt] = __builtin_amdgcn_wmma_f32_16x16x32_bf16(
          false, a.v, false, b.v, (short)0, acc[nt], false, false);
    }
    __syncthreads();
  }

#pragma unroll
  for (int nt = 0; nt < 4; nt++) {
    int col = nbase + nt * 16 + lm;
    float bv = bias[col];
#pragma unroll
    for (int r = 0; r < 8; r++) {
      int row = mbase + wave * 16 + lh * 8 + r;
      Out[(size_t)row * N + col] = bf16_1(acc[nt][r] + bv);
    }
  }
}

// ---------------------------------------------------------------------------
// GEMM 2: Out_f32[M][N] = A_bf16[M][K] @ W_f32[K][N] + bias
// A tile staged with global_load_async_to_lds_b64 (pure bf16 copy), double-buffered.
// ---------------------------------------------------------------------------
__global__ __launch_bounds__(256) void gemm_proj_kernel(
    const uint16_t* __restrict__ A, const float* __restrict__ W,
    const float* __restrict__ bias, float* __restrict__ Out,
    int M, int N, int K) {
  __shared__ uint16_t At[2][128 * LDSTR];
  __shared__ uint16_t Bt[2][64 * LDSTR];

  const int tid  = threadIdx.x;
  const int lane = tid & 31;
  const int wave = tid >> 5;
  const int lm = lane & 15, lh = lane >> 4;
  const int nbase = blockIdx.x * 64;
  const int mbase = blockIdx.y * 128;

  auto fill = [&](int buf, int kk) {
    { // A tile: async bf16 copy, 4 x b64 per thread (row stride 72B, 8B aligned)
      int row = tid >> 1, cs = (tid & 1) * 16;
      const uint16_t* src = A + (size_t)(mbase + row) * K + kk + cs;
      uint16_t* dst = At[buf] + row * LDSTR + cs;
#pragma unroll
      for (int j = 0; j < 4; j++) async_b64(dst + j * 4, src + j * 4);
      __builtin_prefetch(src + 32, 0, 3);
    }
    { // B tile transposed (f32 -> bf16)
      int kr = tid >> 3, ns = (tid & 7) * 8;
      const float* src = W + (size_t)(kk + kr) * N + nbase + ns;
#pragma unroll
      for (int i = 0; i < 8; i++) Bt[buf][(ns + i) * LDSTR + kr] = bf16_1(src[i]);
      __builtin_prefetch(src + (size_t)32 * N, 0, 3);
    }
  };

  v8f acc[4] = {};

  fill(0, 0);
  wait_async0();
  __syncthreads();

  for (int k0 = 0; k0 < K; k0 += 32) {
    const int cur = (k0 >> 5) & 1;
    if (k0 + 32 < K) fill(cur ^ 1, k0 + 32);

    FragB16 a;
    {
      const uint16_t* base = At[cur] + (wave * 16 + lm) * LDSTR;
#pragma unroll
      for (int v = 0; v < 8; v++) a.u[v] = *(const uint32_t*)(base + amap(v, lh));
    }
#pragma unroll
    for (int nt = 0; nt < 4; nt++) {
      FragB16 b;
      const uint16_t* base = Bt[cur] + (nt * 16 + lm) * LDSTR;
#pragma unroll
      for (int v = 0; v < 8; v++) b.u[v] = *(const uint32_t*)(base + bmap(v, lh));
      acc[nt] = __builtin_amdgcn_wmma_f32_16x16x32_bf16(
          false, a.v, false, b.v, (short)0, acc[nt], false, false);
    }
    wait_async0();     // next tile's async copies have landed
    __syncthreads();
  }

#pragma unroll
  for (int nt = 0; nt < 4; nt++) {
    int col = nbase + nt * 16 + lm;
    float bv = bias[col];
#pragma unroll
    for (int r = 0; r < 8; r++) {
      int row = mbase + wave * 16 + lh * 8 + r;
      Out[(size_t)row * N + col] = acc[nt][r] + bv;
    }
  }
}

// ---------------------------------------------------------------------------
// Flash attention: QKV_bf16[B*T][3*D], head h columns: Q @ h*64, K @ 1024+h*64,
// V @ 2048+h*64.  One block = 4 waves = 64 q-rows of one (b,h).
// K tile async-to-LDS, V staged transposed; both double-buffered.
// Output bf16 [B*T][D] with col = h*64+d (already (B,T,D) layout).
// ---------------------------------------------------------------------------
__global__ __launch_bounds__(128) void attn_kernel(
    const uint16_t* __restrict__ QKV, uint16_t* __restrict__ Outbf,
    int T, int D) {
  __shared__ uint16_t Kt[2][32 * KSTR];  // [token 0..31][d 0..63]
  __shared__ uint16_t Vt[2][64 * 34];    // transposed: [d 0..63][token 0..31]
  __shared__ uint16_t Pt[4][16 * 34];    // per-wave P tile [m][k 0..31]

  const int tid  = threadIdx.x;
  const int lane = tid & 31;
  const int wave = tid >> 5;
  const int lm = lane & 15, lh = lane >> 4;

  const int bh = blockIdx.y;       // b*16 + h
  const int b  = bh >> 4;
  const int h  = bh & 15;
  const int qbase  = blockIdx.x * 64;
  const int qrow_w = qbase + wave * 16;
  const size_t rs = (size_t)3 * D;           // 3072 row stride
  const uint16_t* Qp = QKV + (size_t)b * T * rs + h * 64;
  const uint16_t* Kp = Qp + D;
  const uint16_t* Vp = Qp + 2 * D;

  auto stage = [&](int buf, int kb) {
    int kr = tid >> 2, dseg = (tid & 3) * 16;
    const uint16_t* ksrc = Kp + (size_t)(kb + kr) * rs + dseg;
    uint16_t* kdst = Kt[buf] + kr * KSTR + dseg;
#pragma unroll
    for (int j = 0; j < 4; j++) async_b64(kdst + j * 4, ksrc + j * 4);
    __builtin_prefetch(ksrc + 32 * rs, 0, 3);
    const uint16_t* vs = Vp + (size_t)(kb + kr) * rs + dseg;
#pragma unroll
    for (int i = 0; i < 16; i++) Vt[buf][(dseg + i) * 34 + kr] = vs[i];
    __builtin_prefetch(vs + 32 * rs, 0, 3);
  };

  // Q fragments (16x64 -> two 16x32 A-frags), held in registers
  FragB16 qf[2];
  {
    const uint16_t* qr = Qp + (size_t)(qrow_w + lm) * rs;
#pragma unroll
    for (int f = 0; f < 2; f++)
#pragma unroll
      for (int v = 0; v < 8; v++)
        qf[f].u[v] = *(const uint32_t*)(qr + f * 32 + amap(v, lh));
  }

  v8f o[4] = {};
  float m_run[8], l_run[8];
#pragma unroll
  for (int r = 0; r < 8; r++) { m_run[r] = NEG_BIG; l_run[r] = 0.0f; }

  const float scale = 0.125f;  // 1/sqrt(64)
  const int kend = qbase + 64;

  stage(0, 0);
  wait_async0();
  __syncthreads();

  for (int kb = 0; kb < kend; kb += 32) {
    const int cur = (kb >> 5) & 1;
    if (kb + 32 < kend) stage(cur ^ 1, kb + 32);   // overlap with compute

    if (kb <= qrow_w + 15) {  // wave-uniform causal skip
      // ---- S = Q K^T for two 16-token n-tiles
      v8f s0 = {}, s1 = {};
      {
        const uint16_t* kb0 = Kt[cur] + lm * KSTR;
        const uint16_t* kb1 = Kt[cur] + (16 + lm) * KSTR;
        FragB16 f0, f1;
#pragma unroll
        for (int v = 0; v < 8; v++) f0.u[v] = *(const uint32_t*)(kb0 + bmap(v, lh));
#pragma unroll
        for (int v = 0; v < 8; v++) f1.u[v] = *(const uint32_t*)(kb0 + 32 + bmap(v, lh));
        s0 = __builtin_amdgcn_wmma_f32_16x16x32_bf16(false, qf[0].v, false, f0.v, (short)0, s0, false, false);
        s0 = __builtin_amdgcn_wmma_f32_16x16x32_bf16(false, qf[1].v, false, f1.v, (short)0, s0, false, false);
#pragma unroll
        for (int v = 0; v < 8; v++) f0.u[v] = *(const uint32_t*)(kb1 + bmap(v, lh));
#pragma unroll
        for (int v = 0; v < 8; v++) f1.u[v] = *(const uint32_t*)(kb1 + 32 + bmap(v, lh));
        s1 = __builtin_amdgcn_wmma_f32_16x16x32_bf16(false, qf[0].v, false, f0.v, (short)0, s1, false, false);
        s1 = __builtin_amdgcn_wmma_f32_16x16x32_bf16(false, qf[1].v, false, f1.v, (short)0, s1, false, false);
      }

      // ---- online softmax over the 32 new columns
      uint16_t* pw = Pt[wave];
#pragma unroll
      for (int r = 0; r < 8; r++) {
        int qg = qrow_w + lh * 8 + r;
        float e0 = s0[r] * scale, e1 = s1[r] * scale;
        e0 = (kb + lm      <= qg) ? e0 : NEG_BIG;
        e1 = (kb + 16 + lm <= qg) ? e1 : NEG_BIG;
        float mx = fmaxf(e0, e1);
        mx = fmaxf(mx, __shfl_xor(mx, 1, 32));
        mx = fmaxf(mx, __shfl_xor(mx, 2, 32));
        mx = fmaxf(mx, __shfl_xor(mx, 4, 32));
        mx = fmaxf(mx, __shfl_xor(mx, 8, 32));
        float mN = fmaxf(m_run[r], mx);
        float corr = __expf(m_run[r] - mN);
        float p0 = __expf(e0 - mN), p1 = __expf(e1 - mN);
        float sum = p0 + p1;
        sum += __shfl_xor(sum, 1, 32);
        sum += __shfl_xor(sum, 2, 32);
        sum += __shfl_xor(sum, 4, 32);
        sum += __shfl_xor(sum, 8, 32);
        l_run[r] = l_run[r] * corr + sum;
        m_run[r] = mN;
        o[0][r] *= corr; o[1][r] *= corr; o[2][r] *= corr; o[3][r] *= corr;
        int mrow = lh * 8 + r;
        pw[mrow * 34 + lm]      = bf16_1(p0);   // C-layout -> A-layout via LDS
        pw[mrow * 34 + 16 + lm] = bf16_1(p1);
      }
      asm volatile("s_wait_dscnt 0" ::: "memory"); // wave-local LDS RAW fence

      // ---- O += P (16x32, A-frag) x V (32x64, four B-frags)
      FragB16 pf;
#pragma unroll
      for (int v = 0; v < 8; v++)
        pf.u[v] = *(const uint32_t*)(pw + lm * 34 + amap(v, lh));
#pragma unroll
      for (int nt = 0; nt < 4; nt++) {
        FragB16 vf;
        const uint16_t* vb = Vt[cur] + (nt * 16 + lm) * 34;
#pragma unroll
        for (int v = 0; v < 8; v++) vf.u[v] = *(const uint32_t*)(vb + bmap(v, lh));
        o[nt] = __builtin_amdgcn_wmma_f32_16x16x32_bf16(
            false, pf.v, false, vf.v, (short)0, o[nt], false, false);
      }
    }
    wait_async0();
    __syncthreads();
  }

  // epilogue: normalize and emit bf16 at (B,T,D) position
#pragma unroll
  for (int nt = 0; nt < 4; nt++) {
#pragma unroll
    for (int r = 0; r < 8; r++) {
      int row = qrow_w + lh * 8 + r;
      float val = o[nt][r] * (1.0f / l_run[r]);
      size_t idx = ((size_t)b * T + row) * (size_t)D + h * 64 + nt * 16 + lm;
      Outbf[idx] = bf16_1(val);
    }
  }
}

// ---------------------------------------------------------------------------
extern "C" void kernel_launch(void* const* d_in, const int* in_sizes, int n_in,
                              void* d_out, int out_size, void* d_ws, size_t ws_size,
                              hipStream_t stream) {
  const float* x      = (const float*)d_in[0];
  const float* w_qkv  = (const float*)d_in[1];
  const float* b_qkv  = (const float*)d_in[2];
  const float* w_proj = (const float*)d_in[3];
  const float* b_proj = (const float*)d_in[4];
  float* out = (float*)d_out;

  const int D = 1024, T = 2048;
  const int B = in_sizes[0] / (T * D);     // = 4
  const int M = B * T;                     // 8192

  uint16_t* qkv_bf  = (uint16_t*)d_ws;                // [M][3D] bf16 (48 MB)
  uint16_t* attn_bf = qkv_bf + (size_t)M * 3 * D;     // [M][D]  bf16 (16 MB)

  gemm_qkv_kernel<<<dim3((3 * D) / 64, M / 128), dim3(256), 0, stream>>>(
      x, w_qkv, b_qkv, qkv_bf, M, 3 * D, D);
  attn_kernel<<<dim3(T / 64, B * 16), dim3(128), 0, stream>>>(qkv_bf, attn_bf, T, D);
  gemm_proj_kernel<<<dim3(D / 64, M / 128), dim3(256), 0, stream>>>(
      attn_bf, w_proj, b_proj, out, M, D, D);
}
